// _NonLocalBlockND_43173011259773
// MI455X (gfx1250) — matrix operand order repaired and checked by reading, not compiled, not run on hardware
//
#include <hip/hip_runtime.h>
#include <cstddef>

// ---------------------------------------------------------------------------
// NonLocal block on MI455X (gfx1250, wave32, WMMA bf16).
// Flash-attention tiling: f [B,4096,4096] is never materialized.
// All GEMMs use v_wmma_f32_16x16x32_bf16 (f32 accumulate).
// Fragment loads are batched ahead of WMMA groups so loadcnt waits stagger
// instead of serializing (load -> wait 0 -> wmma).
// Workspace: theta/phi [B,N,Ci], g [B,Ci,N], o [B,N,Ci] bf16 (16MB)
//            + pre-converted bf16 weights (256KB).
// ---------------------------------------------------------------------------

#define B_ 4
#define C_ 256
#define CI_ 128
#define N_ 4096

typedef __bf16 bf16;
typedef __bf16 v16bf __attribute__((ext_vector_type(16)));
typedef __bf16 bf16x8 __attribute__((ext_vector_type(8)));
typedef float  v8f   __attribute__((ext_vector_type(8)));

union Frag {                 // one WMMA 16-bit A/B fragment (16 bf16 per lane)
    v16bf  bf;
    bf16x8 h[2];
    bf16   e[16];
};

static __device__ __forceinline__ v8f v8zero() {
    v8f v;
    #pragma unroll
    for (int i = 0; i < 8; ++i) v[i] = 0.0f;
    return v;
}

static __device__ __forceinline__ v8f wmma_bf16(const Frag& a, const Frag& b, v8f c) {
    return __builtin_amdgcn_wmma_f32_16x16x32_bf16(
        /*neg_a=*/false, a.bf, /*neg_b=*/false, b.bf,
        /*c_mod=*/(short)0, c, /*reuse_a=*/false, /*reuse_b=*/false);
}

// ---------------------------------------------------------------------------
// One-shot f32 -> bf16 weight conversion.
// ---------------------------------------------------------------------------
__global__ __launch_bounds__(256) void wcvt_kernel(
    const float* __restrict__ src, bf16* __restrict__ dst, int n)
{
    int i = (blockIdx.x * 256 + threadIdx.x) * 2;
    if (i < n) {
        dst[i]     = (bf16)src[i];       // pairs encourage v_cvt_pk_bf16_f32
        dst[i + 1] = (bf16)src[i + 1];
    }
}

// ---------------------------------------------------------------------------
// Projection: out[o,n] = sum_c w[o,c] * in[c,n] + bias[o]   (per batch)
// M = n (64 rows per block, 16 per wave), N = o (Ci=128), K = c (C=256).
// transposed==1 -> out is [B,N,Ci] bf16 ; transposed==0 -> out is [B,Ci,N] bf16
// ---------------------------------------------------------------------------
__global__ __launch_bounds__(128) void proj_kernel(
    const float* __restrict__ in,     // [B, C, N] f32
    const bf16*  __restrict__ w,      // [Ci, C]   bf16 (pre-converted)
    const float* __restrict__ bias,   // [Ci]      f32
    bf16* __restrict__ out,
    int transposed)
{
    __shared__ __align__(16) bf16 xT[64][32];   // input chunk, transposed, bf16

    const int b      = blockIdx.y;
    const int n0     = blockIdx.x * 64;
    const int tid    = threadIdx.x;
    const int lane   = tid & 31;
    const int wv     = tid >> 5;
    const int lane16 = lane & 15;
    const int half   = lane >> 4;

    v8f acc[8];
    #pragma unroll
    for (int c = 0; c < 8; ++c) acc[c] = v8zero();

    const float* xb = in + (size_t)b * C_ * N_;

    for (int kc = 0; kc < C_ / 32; ++kc) {
        __syncthreads();
        // stage 32 x 64 f32 chunk of input, transposed, as bf16 into LDS
        for (int idx = tid; idx < 32 * 64; idx += 128) {
            int cc = idx >> 6, n = idx & 63;
            xT[n][cc] = (bf16)xb[(size_t)(kc * 32 + cc) * N_ + n0 + n];
        }
        __syncthreads();

        // batch all fragment loads first, then the 8 WMMAs
        Frag afr;
        afr.h[0] = *(const bf16x8*)&xT[wv * 16 + lane16][8 * half];
        afr.h[1] = *(const bf16x8*)&xT[wv * 16 + lane16][16 + 8 * half];

        Frag bfr[8];
        #pragma unroll
        for (int c = 0; c < 8; ++c) {
            const bf16* wrow = w + (size_t)(16 * c + lane16) * C_ + kc * 32 + 8 * half;
            bfr[c].h[0] = *(const bf16x8*)(wrow);
            bfr[c].h[1] = *(const bf16x8*)(wrow + 16);
        }
        #pragma unroll
        for (int c = 0; c < 8; ++c)
            acc[c] = wmma_bf16(afr, bfr[c], acc[c]);
    }

    // bias + store
    #pragma unroll
    for (int c = 0; c < 8; ++c) {
        const int o_col = 16 * c + lane16;
        const float bb = bias[o_col];
        if (transposed) {                                   // [B, N, Ci]
            #pragma unroll
            for (int r = 0; r < 8; ++r) {
                int n = n0 + wv * 16 + r + 8 * half;
                out[((size_t)b * N_ + n) * CI_ + o_col] = (bf16)(acc[c][r] + bb);
            }
        } else {                                            // [B, Ci, N]
            bf16x8 tv;
            #pragma unroll
            for (int r = 0; r < 8; ++r) tv[r] = (bf16)(acc[c][r] + bb);
            int n = n0 + wv * 16 + 8 * half;                // rows contiguous in n
            *(bf16x8*)&out[((size_t)b * CI_ + o_col) * N_ + n] = tv;
        }
    }
}

// ---------------------------------------------------------------------------
// Flash attention: per wave, 16 query rows; loop over keys in steps of 32.
//   S = theta[16 x 128] @ phi^T      -> 8 WMMA (phi frags batch-loaded)
//   g fragments loaded BEFORE softmax so they fly during exp/shuffle work
//   online softmax (row state m,l)   -> shfl_xor reductions within 16-lane half
//   O += P[16 x 32] @ g[32 x 128]    -> 8 WMMA (P routed through LDS to A-layout)
// ---------------------------------------------------------------------------
__global__ __launch_bounds__(128) void attn_kernel(
    const bf16* __restrict__ theta,  // [B, N, Ci]
    const bf16* __restrict__ phi,    // [B, N, Ci]
    const bf16* __restrict__ g,      // [B, Ci, N]
    bf16* __restrict__ o)            // [B, N, Ci]
{
    __shared__ __align__(16) bf16 P_lds[4][16][32];  // per-wave P tile

    const int b      = blockIdx.y;
    const int lane   = threadIdx.x & 31;
    const int wv     = threadIdx.x >> 5;
    const int lane16 = lane & 15;
    const int half   = lane >> 4;
    const int qbase  = blockIdx.x * 64 + wv * 16;

    const bf16* th = theta + (size_t)b * N_ * CI_;
    const bf16* ph = phi   + (size_t)b * N_ * CI_;
    const bf16* gz = g     + (size_t)b * CI_ * N_;

    // A fragments of theta (persist across the whole key loop)
    Frag afr[4];
    {
        const bf16* row = th + (size_t)(qbase + lane16) * CI_;
        #pragma unroll
        for (int kc = 0; kc < 4; ++kc) {
            afr[kc].h[0] = *(const bf16x8*)(row + 32 * kc + 8 * half);
            afr[kc].h[1] = *(const bf16x8*)(row + 32 * kc + 16 + 8 * half);
        }
    }

    v8f acc[8];
    #pragma unroll
    for (int c = 0; c < 8; ++c) acc[c] = v8zero();
    float mrow[8], lrow[8];
    #pragma unroll
    for (int r = 0; r < 8; ++r) { mrow[r] = -1e30f; lrow[r] = 0.0f; }

    for (int kt = 0; kt < N_ / 32; ++kt) {
        const int kbase = kt * 32;

        // prefetch next key tile (speculative, fault-free; lands in workspace)
        __builtin_prefetch(ph + (size_t)(kbase + 32 + lane16) * CI_, 0, 1);
        __builtin_prefetch(gz + (size_t)lane16 * N_ + kbase + 32, 0, 1);

        // ---- batch-load all 8 phi B fragments, then 8 score WMMAs ----
        Frag pbf[8];                                  // [t][kc] flattened
        #pragma unroll
        for (int t = 0; t < 2; ++t) {
            const bf16* prow = ph + (size_t)(kbase + t * 16 + lane16) * CI_;
            #pragma unroll
            for (int kc = 0; kc < 4; ++kc) {
                pbf[t * 4 + kc].h[0] = *(const bf16x8*)(prow + 32 * kc + 8 * half);
                pbf[t * 4 + kc].h[1] = *(const bf16x8*)(prow + 32 * kc + 8 * half + 16);
            }
        }
        v8f S[2];
        #pragma unroll
        for (int t = 0; t < 2; ++t) {
            v8f s = v8zero();
            #pragma unroll
            for (int kc = 0; kc < 4; ++kc)
                s = wmma_bf16(afr[kc], pbf[t * 4 + kc], s);
            S[t] = s;
        }

        // ---- issue g fragment loads NOW (before the asm memory clobbers),
        //      so they overlap the whole softmax latency ----
        Frag gfr[8];
        #pragma unroll
        for (int c = 0; c < 8; ++c) {
            const bf16* gcol =
                gz + (size_t)(16 * c + lane16) * N_ + kbase + 8 * half;
            gfr[c].h[0] = *(const bf16x8*)(gcol);
            gfr[c].h[1] = *(const bf16x8*)(gcol + 16);
        }

        // ---- online softmax (rows r + 8*half, cols striped over 16 lanes) ----
        float newm[8];
        #pragma unroll
        for (int r = 0; r < 8; ++r) {
            float v = fmaxf(S[0][r], S[1][r]);
            v = fmaxf(v, __shfl_xor(v, 1, 32));
            v = fmaxf(v, __shfl_xor(v, 2, 32));
            v = fmaxf(v, __shfl_xor(v, 4, 32));
            v = fmaxf(v, __shfl_xor(v, 8, 32));
            newm[r] = fmaxf(mrow[r], v);
        }
        #pragma unroll
        for (int r = 0; r < 8; ++r) {
            float p0 = __expf(S[0][r] - newm[r]);
            float p1 = __expf(S[1][r] - newm[r]);
            S[0][r] = p0; S[1][r] = p1;
            float rs = p0 + p1;
            rs += __shfl_xor(rs, 1, 32);
            rs += __shfl_xor(rs, 2, 32);
            rs += __shfl_xor(rs, 4, 32);
            rs += __shfl_xor(rs, 8, 32);
            const float scale = __expf(mrow[r] - newm[r]);
            lrow[r] = lrow[r] * scale + rs;
            mrow[r] = newm[r];
            #pragma unroll
            for (int c = 0; c < 8; ++c) acc[c][r] *= scale;
        }

        // ---- P: C-tile layout -> A-fragment layout via per-wave LDS tile ----
        #pragma unroll
        for (int t = 0; t < 2; ++t)
            #pragma unroll
            for (int r = 0; r < 8; ++r)
                P_lds[wv][r + 8 * half][t * 16 + lane16] = (bf16)S[t][r];
        asm volatile("s_wait_dscnt 0x0" ::: "memory");   // DS ops in-order per wave
        Frag pfr;
        pfr.h[0] = *(const bf16x8*)&P_lds[wv][lane16][8 * half];
        pfr.h[1] = *(const bf16x8*)&P_lds[wv][lane16][16 + 8 * half];
        asm volatile("s_wait_dscnt 0x0" ::: "memory");

        // ---- O += P @ g with pre-loaded g fragments ----
        #pragma unroll
        for (int c = 0; c < 8; ++c)
            acc[c] = wmma_bf16(pfr, gfr[c], acc[c]);
    }

    // ---- normalize and store o as [B, N, Ci] bf16 ----
    bf16* ob = o + (size_t)b * N_ * CI_;
    #pragma unroll
    for (int c = 0; c < 8; ++c) {
        const int ci = 16 * c + lane16;
        #pragma unroll
        for (int r = 0; r < 8; ++r) {
            const int n = qbase + r + 8 * half;
            ob[(size_t)n * CI_ + ci] = (bf16)(acc[c][r] / lrow[r]);
        }
    }
}

// ---------------------------------------------------------------------------
// Output projection + BatchNorm(eval) + residual:
//   out[co,n] = BN( sum_ci w_w[co,ci]*o[n,ci] + w_b[co] ) + z[co,n]
// M = co (16 per wave), N = n (64 per block), K = ci (128).
// ---------------------------------------------------------------------------
__global__ __launch_bounds__(128) void outproj_kernel(
    const bf16*  __restrict__ o,      // [B, N, Ci] bf16
    const bf16*  __restrict__ ww,     // [C, Ci] bf16 (pre-converted)
    const float* __restrict__ wb,     // [C]
    const float* __restrict__ gamma,
    const float* __restrict__ beta,
    const float* __restrict__ mean,
    const float* __restrict__ var,
    const float* __restrict__ z,      // [B, C, N]
    float* __restrict__ outp)         // [B, C, N]
{
    const int b      = blockIdx.z;
    const int co0    = blockIdx.y * 64;
    const int n0     = blockIdx.x * 64;
    const int lane   = threadIdx.x & 31;
    const int wv     = threadIdx.x >> 5;
    const int lane16 = lane & 15;
    const int half   = lane >> 4;
    const int cobase = co0 + wv * 16;

    v8f acc[4];
    #pragma unroll
    for (int t = 0; t < 4; ++t) acc[t] = v8zero();

    const bf16* ob = o + (size_t)b * N_ * CI_;

    #pragma unroll
    for (int kc = 0; kc < 4; ++kc) {
        // batch loads: A fragment + all 4 B fragments, then 4 WMMAs
        Frag afr;
        const bf16* wrow = ww + (size_t)(cobase + lane16) * CI_ + 32 * kc;
        afr.h[0] = *(const bf16x8*)(wrow + 8 * half);
        afr.h[1] = *(const bf16x8*)(wrow + 16 + 8 * half);

        Frag bfr[4];
        #pragma unroll
        for (int t = 0; t < 4; ++t) {
            const bf16* orow =
                ob + (size_t)(n0 + t * 16 + lane16) * CI_ + 32 * kc + 8 * half;
            bfr[t].h[0] = *(const bf16x8*)(orow);
            bfr[t].h[1] = *(const bf16x8*)(orow + 16);
        }
        #pragma unroll
        for (int t = 0; t < 4; ++t)
            acc[t] = wmma_bf16(afr, bfr[t], acc[t]);
    }

    // per-row (co) BN parameters
    float biasr[8], invr[8], betr[8], meanr[8];
    #pragma unroll
    for (int r = 0; r < 8; ++r) {
        const int co = cobase + r + 8 * half;
        biasr[r] = wb[co];
        invr[r]  = gamma[co] * rsqrtf(var[co] + 1e-5f);
        betr[r]  = beta[co];
        meanr[r] = mean[co];
    }

    #pragma unroll
    for (int t = 0; t < 4; ++t) {
        const int n = n0 + t * 16 + lane16;
        #pragma unroll
        for (int r = 0; r < 8; ++r) {
            const int co = cobase + r + 8 * half;
            const size_t idx = ((size_t)b * C_ + co) * N_ + n;
            outp[idx] = (acc[t][r] + biasr[r] - meanr[r]) * invr[r] + betr[r] + z[idx];
        }
    }
}

// ---------------------------------------------------------------------------
extern "C" void kernel_launch(void* const* d_in, const int* in_sizes, int n_in,
                              void* d_out, int out_size, void* d_ws, size_t ws_size,
                              hipStream_t stream) {
    (void)in_sizes; (void)n_in; (void)out_size; (void)ws_size;

    const float* x       = (const float*)d_in[0];
    const float* y       = (const float*)d_in[1];
    const float* z       = (const float*)d_in[2];
    const float* theta_w = (const float*)d_in[3];
    const float* theta_b = (const float*)d_in[4];
    const float* phi_w   = (const float*)d_in[5];
    const float* phi_b   = (const float*)d_in[6];
    const float* g_w     = (const float*)d_in[7];
    const float* g_b     = (const float*)d_in[8];
    const float* w_w     = (const float*)d_in[9];
    const float* w_b     = (const float*)d_in[10];
    const float* bn_g    = (const float*)d_in[11];
    const float* bn_b    = (const float*)d_in[12];
    const float* bn_m    = (const float*)d_in[13];
    const float* bn_v    = (const float*)d_in[14];
    float* out = (float*)d_out;

    const size_t pe = (size_t)B_ * N_ * CI_;   // 2M elements per activation buf
    const int    we = CI_ * C_;                // 32768 elements per weight
    bf16* theta = (bf16*)d_ws;                 // [B,N,Ci]
    bf16* phi   = theta + pe;                  // [B,N,Ci]
    bf16* gws   = phi + pe;                    // [B,Ci,N]
    bf16* ows   = gws + pe;                    // [B,N,Ci]
    bf16* twbf  = ows + pe;                    // [Ci,C]
    bf16* pwbf  = twbf + we;                   // [Ci,C]
    bf16* gwbf  = pwbf + we;                   // [Ci,C]
    bf16* wwbf  = gwbf + we;                   // [C,Ci]

    const dim3 blk(128);

    // pre-convert weights to bf16 (hardware cvt, one pass, 256KB total)
    wcvt_kernel<<<we / 512, 256, 0, stream>>>(theta_w, twbf, we);
    wcvt_kernel<<<we / 512, 256, 0, stream>>>(phi_w,   pwbf, we);
    wcvt_kernel<<<we / 512, 256, 0, stream>>>(g_w,     gwbf, we);
    wcvt_kernel<<<we / 512, 256, 0, stream>>>(w_w,     wwbf, we);

    const dim3 pgrid(N_ / 64, B_);
    proj_kernel<<<pgrid, blk, 0, stream>>>(x, twbf, theta_b, theta, 1);
    proj_kernel<<<pgrid, blk, 0, stream>>>(y, pwbf, phi_b,   phi,   1);
    proj_kernel<<<pgrid, blk, 0, stream>>>(z, gwbf, g_b,     gws,   0);

    attn_kernel<<<dim3(N_ / 64, B_), blk, 0, stream>>>(theta, phi, gws, ows);

    outproj_kernel<<<dim3(N_ / 64, C_ / 64, B_), blk, 0, stream>>>(
        ows, wwbf, w_b, bn_g, bn_b, bn_m, bn_v, z, out);
}
